// StoryMemoryModel_64201171140754
// MI455X (gfx1250) — compile-verified, matrix operand order checked by reading, not run on hardware
//
#include <hip/hip_runtime.h>
#include <hip/hip_bf16.h>
#include <math.h>

// ---- model dims (fixed by reference) ----
#define DV     512
#define TV     2048
#define BV     2
#define LV     6
#define FFV    2048
#define SV     8
#define DSV    32
#define KCONV  15
#define VVOC   32000
#define MROWS  (BV*TV)   // 4096 token rows

typedef __attribute__((ext_vector_type(16))) __bf16 v16bf;
typedef __attribute__((ext_vector_type(8)))  __bf16 v8bf;
typedef __attribute__((ext_vector_type(8)))  float  v8f;

static __device__ __forceinline__ v16bf ld_frag(const __bf16* p) {
  v8bf lo = *(const v8bf*)p;
  v8bf hi = *(const v8bf*)(p + 16);
  return __builtin_shufflevector(lo, hi, 0,1,2,3,4,5,6,7,8,9,10,11,12,13,14,15);
}

#define WMMA_BF16(a, b, c) __builtin_amdgcn_wmma_f32_16x16x32_bf16( \
    false, (a), false, (b), (short)0, (c), false, false)

// ---------------------------------------------------------------------------
// Register-tiled, software-pipelined GEMM:
//   O[M,N] (=|+=) A_bf16[M,Kd] * W_bf16[N,Kd]^T   (both K-major).
// Block = 8 waves = 128M x 128N; wave = 32M x 64N (2 M-frags x 4 N-frags,
// 8 x V_WMMA_F32_16X16X32_BF16 per 32-wide K-step, fp32 accumulate).
// K loop is unrolled by 2 with role-alternating ping-pong fragment sets, so
// loads always lead the WMMAs by one K-step and the hand-off needs no
// register copies (loads write directly into the loop-carried set).
// Requires Kd/32 even (holds for all K used here: 256, 512, 2048).
// A-frag (16-bit 16x32, ISA 7.12.2): lane = half*16+r, row M=r,
//   K in {k0..k0+7, k0+16..k0+23}, k0 = kk + half*8. B mirrored (K-major W).
// C/D: acc[i] -> (M = half*8 + i, N = r).
// ---------------------------------------------------------------------------
template<int MODE>  // 0 = store f32, 1 = add into f32, 2 = nontemporal store
__global__ __launch_bounds__(256)
void k_gemm_bf16(const __bf16* __restrict__ A, const __bf16* __restrict__ W,
                 float* __restrict__ O, int N, int Kd) {
  const int lane   = threadIdx.x & 31;
  const int wave   = threadIdx.x >> 5;
  const int mgroup = wave & 3;        // 4 M-subtiles of 32 rows
  const int ngroup = wave >> 2;       // 2 N-subtiles of 64 cols
  const int half   = lane >> 4;
  const int r      = lane & 15;

  const int mb = blockIdx.x * 128 + mgroup * 32;
  const int nb = blockIdx.y * 128 + ngroup * 64;
  if (nb >= N || mb >= MROWS) return;

  const __bf16* a0 = A + (size_t)(mb + r) * Kd + half * 8;
  const __bf16* a1 = A + (size_t)(mb + 16 + r) * Kd + half * 8;
  const __bf16* w0 = W + (size_t)(nb + r) * Kd + half * 8;
  const size_t wstep = (size_t)16 * Kd;

  const int ksteps = Kd >> 5;   // even for all K used here

  // c-set: fragments for the even K-step (loop-carried, loads write into it)
  v16bf ca0 = ld_frag(a0);
  v16bf ca1 = ld_frag(a1);
  v16bf cb[4];
#pragma unroll
  for (int j = 0; j < 4; ++j) cb[j] = ld_frag(w0 + (size_t)j * wstep);

  v8f acc[2][4] = {};
  for (int ks = 0; ks < ksteps; ks += 2) {
    // n-set: fragments for the odd K-step ks+1 (fresh values every iteration)
    const int k1 = (ks + 1) * 32;
    v16bf na0 = ld_frag(a0 + k1);
    v16bf na1 = ld_frag(a1 + k1);
    v16bf nb[4];
#pragma unroll
    for (int j = 0; j < 4; ++j) nb[j] = ld_frag(w0 + (size_t)j * wstep + k1);

    // 8 WMMAs on the c-set (step ks)
#pragma unroll
    for (int j = 0; j < 4; ++j) {
      acc[0][j] = WMMA_BF16(ca0, cb[j], acc[0][j]);
      acc[1][j] = WMMA_BF16(ca1, cb[j], acc[1][j]);
    }

    // refill c-set for step ks+2 (loads write directly into carried regs)
    if (ks + 2 < ksteps) {
      const int k2 = (ks + 2) * 32;
      ca0 = ld_frag(a0 + k2);
      ca1 = ld_frag(a1 + k2);
#pragma unroll
      for (int j = 0; j < 4; ++j) cb[j] = ld_frag(w0 + (size_t)j * wstep + k2);
    }

    // 8 WMMAs on the n-set (step ks+1)
#pragma unroll
    for (int j = 0; j < 4; ++j) {
      acc[0][j] = WMMA_BF16(na0, nb[j], acc[0][j]);
      acc[1][j] = WMMA_BF16(na1, nb[j], acc[1][j]);
    }
  }

#pragma unroll
  for (int mi = 0; mi < 2; ++mi) {
#pragma unroll
    for (int j = 0; j < 4; ++j) {
      size_t base = (size_t)(mb + mi * 16 + half * 8) * (size_t)N + nb + j * 16 + r;
#pragma unroll
      for (int i = 0; i < 8; ++i) {
        size_t idx = base + (size_t)i * N;
        if (MODE == 0)      O[idx] = acc[mi][j][i];
        else if (MODE == 1) O[idx] += acc[mi][j][i];
        else                __builtin_nontemporal_store(acc[mi][j][i], &O[idx]);
      }
    }
  }
}

// f32 -> bf16 bulk convert (4 elems/thread); all sizes are multiples of 4
__global__ void k_cvt_bf16(const float* __restrict__ in, __bf16* __restrict__ o, int n4) {
  int i = blockIdx.x * blockDim.x + threadIdx.x;
  if (i >= n4) return;
  const float4 f = ((const float4*)in)[i];
  o[4 * i + 0] = (__bf16)f.x;
  o[4 * i + 1] = (__bf16)f.y;
  o[4 * i + 2] = (__bf16)f.z;
  o[4 * i + 3] = (__bf16)f.w;
}

// ---------------------------------------------------------------------------
// RMSNorm over D=512 (one block per row, 256 threads x 2 elems), optional
// token-gather on the input row; writes f32 and/or bf16 (either may be null).
// ---------------------------------------------------------------------------
__global__ __launch_bounds__(256)
void k_rmsnorm(const float* __restrict__ in, const int* __restrict__ tokens,
               const float* __restrict__ w, float* __restrict__ of32,
               __bf16* __restrict__ obf) {
  const int row = blockIdx.x;
  const int src = tokens ? tokens[row] : row;
  const float* xr = in + (size_t)src * DV;
  const int tid = threadIdx.x;

  float x0 = xr[tid], x1 = xr[tid + 256];
  float ss = x0 * x0 + x1 * x1;
#pragma unroll
  for (int d = 16; d > 0; d >>= 1) ss += __shfl_down(ss, d, 32);

  __shared__ float wsum[8];
  __shared__ float scale_s;
  if ((tid & 31) == 0) wsum[tid >> 5] = ss;
  __syncthreads();
  if (tid == 0) {
    float tot = 0.f;
#pragma unroll
    for (int i = 0; i < 8; ++i) tot += wsum[i];
    scale_s = rsqrtf(tot / (float)DV + 1e-6f);
  }
  __syncthreads();
  const float sc = scale_s;
  float y0 = x0 * sc * w[tid];
  float y1 = x1 * sc * w[tid + 256];
  size_t o = (size_t)row * DV + tid;
  if (of32) { of32[o] = y0; of32[o + 256] = y1; }
  if (obf)  { obf[o] = (__bf16)y0; obf[o + 256] = (__bf16)y1; }
}

// ---------------------------------------------------------------------------
// Causal depthwise conv (K=15) fused with sigmoid(gate) -> bf16 GEMM operand.
// gv layout: [m, 2D], gate = [:,0:512], val = [:,512:1024].
// ---------------------------------------------------------------------------
__global__ void k_convgate(const float* __restrict__ gv, const float* __restrict__ cw,
                           __bf16* __restrict__ cg) {
  int idx = blockIdx.x * blockDim.x + threadIdx.x;
  if (idx >= MROWS * DV) return;
  const int d = idx % DV;
  const int m = idx / DV;
  const int t = m % TV;

  float acc = 0.f;
  const int kstart = (t >= KCONV - 1) ? 0 : (KCONV - 1 - t);
#pragma unroll
  for (int k = 0; k < KCONV; ++k) {
    if (k >= kstart) {
      int mm = m - (KCONV - 1) + k;   // same batch (kstart guards t boundary)
      acc += cw[d * KCONV + k] * gv[(size_t)mm * (2 * DV) + DV + d];
    }
  }
  float gate = gv[(size_t)m * (2 * DV) + d];
  float sg = 1.f / (1.f + __expf(-gate));
  cg[idx] = (__bf16)(acc * sg);
}

// ---------------------------------------------------------------------------
// Per-row small projections: g = sigmoid(hm*wg^T) (S=8), v = hm*wv^T (DS=32),
// rw = softmax(hm*rq^T) (S=8). One block per row. The activation row is
// staged into LDS with gfx1250 async-to-LDS (ASYNCcnt).
// ---------------------------------------------------------------------------
__global__ __launch_bounds__(256)
void k_smalldense(const float* __restrict__ hm, const float* __restrict__ wg,
                  const float* __restrict__ wv, const float* __restrict__ rq,
                  float* __restrict__ gout, float* __restrict__ vout,
                  float* __restrict__ rwout) {
  __shared__ float xs[DV];
  __shared__ float outs[48];
  const int row = blockIdx.x;
  const int tid = threadIdx.x;

  {
    const float* g0 = hm + (size_t)row * DV + tid;
    const float* g1 = g0 + 256;
    unsigned l0 = (unsigned)(unsigned long long)&xs[tid];
    unsigned l1 = (unsigned)(unsigned long long)&xs[tid + 256];
    asm volatile("global_load_async_to_lds_b32 %0, %1, off" :: "v"(l0), "v"(g0) : "memory");
    asm volatile("global_load_async_to_lds_b32 %0, %1, off" :: "v"(l1), "v"(g1) : "memory");
    asm volatile("s_wait_asynccnt 0x0" ::: "memory");
  }
  __syncthreads();

  if (tid < 48) {
    const float* wr = (tid < 8)  ? (wg + (size_t)tid * DV)
                    : (tid < 40) ? (wv + (size_t)(tid - 8) * DV)
                                 : (rq + (size_t)(tid - 40) * DV);
    float acc = 0.f;
    for (int d = 0; d < DV; ++d) acc += xs[d] * wr[d];
    outs[tid] = acc;
  }
  __syncthreads();
  if (tid < 8) {
    gout[(size_t)row * SV + tid] = 1.f / (1.f + __expf(-outs[tid]));
  } else if (tid < 40) {
    vout[(size_t)row * DSV + (tid - 8)] = outs[tid];
  } else if (tid == 40) {
    float mx = outs[40];
#pragma unroll
    for (int i = 41; i < 48; ++i) mx = fmaxf(mx, outs[i]);
    float e[8], sum = 0.f;
#pragma unroll
    for (int i = 0; i < 8; ++i) { e[i] = __expf(outs[40 + i] - mx); sum += e[i]; }
#pragma unroll
    for (int i = 0; i < 8; ++i) rwout[(size_t)row * SV + i] = e[i] / sum;
  }
}

// ---------------------------------------------------------------------------
// Affine associative scan over T per (b,s,ds): m_t = (1-g)*m_{t-1} + g*v.
// One wave per chain; 64-step chunks per lane; cross-lane affine compose via
// shfl_up; replay with exclusive carry. Fuses rf = rw*mem -> bf16.
// ---------------------------------------------------------------------------
__global__ __launch_bounds__(256)
void k_scan(const float* __restrict__ gbuf, const float* __restrict__ vbuf,
            const float* __restrict__ rwbuf, const float* __restrict__ memin,
            float* __restrict__ memout, __bf16* __restrict__ rf) {
  const int wave  = threadIdx.x >> 5;
  const int lane  = threadIdx.x & 31;
  const int chain = blockIdx.x * 8 + wave;          // 0 .. B*S*DS-1 = 511
  const int b  = chain / (SV * DSV);
  const int s  = (chain / DSV) % SV;
  const int ds = chain % DSV;
  const int C  = TV / 32;                           // 64 steps per lane
  const int t0 = lane * C;
  const size_t rowbase = (size_t)b * TV;

  float A = 1.f, Bc = 0.f;
  for (int i = 0; i < C; ++i) {
    const size_t t = rowbase + t0 + i;
    float g  = gbuf[t * SV + s];
    float bv = g * vbuf[t * DSV + ds];
    float a  = 1.f - g;
    A = a * A;
    Bc = a * Bc + bv;
  }
  for (int d = 1; d < 32; d <<= 1) {
    float Al = __shfl_up(A, d, 32);
    float Bl = __shfl_up(Bc, d, 32);
    if (lane >= d) { Bc = A * Bl + Bc; A = A * Al; }
  }
  float Ae = __shfl_up(A, 1, 32);
  float Be = __shfl_up(Bc, 1, 32);
  if (lane == 0) { Ae = 1.f; Be = 0.f; }
  float m = Ae * memin[((size_t)b * SV + s) * DSV + ds] + Be;

  for (int i = 0; i < C; ++i) {
    const size_t t = rowbase + t0 + i;
    float g  = gbuf[t * SV + s];
    float bv = g * vbuf[t * DSV + ds];
    m = (1.f - g) * m + bv;
    float rw = rwbuf[t * SV + s];
    rf[t * (SV * DSV) + s * DSV + ds] = (__bf16)(rw * m);
  }
  if (lane == 31) memout[((size_t)b * SV + s) * DSV + ds] = m;
}

// SiLU(gf)*uf -> bf16 operand for the Wof GEMM
__global__ void k_siluMul(const float* __restrict__ gf, const float* __restrict__ uf,
                          __bf16* __restrict__ up, int n) {
  int i = blockIdx.x * blockDim.x + threadIdx.x;
  if (i >= n) return;
  float g = gf[i];
  up[i] = (__bf16)((g / (1.f + __expf(-g))) * uf[i]);
}

// ---------------------------------------------------------------------------
extern "C" void kernel_launch(void* const* d_in, const int* in_sizes, int n_in,
                              void* d_out, int out_size, void* d_ws, size_t ws_size,
                              hipStream_t stream) {
  const int*   tokens  = (const int*)  d_in[0];
  const float* embed   = (const float*)d_in[1];
  const float* ln_in   = (const float*)d_in[2];
  const float* ln1     = (const float*)d_in[3];
  const float* mup     = (const float*)d_in[4];
  const float* convw   = (const float*)d_in[5];
  const float* mdown   = (const float*)d_in[6];
  const float* lnmem   = (const float*)d_in[7];
  const float* wg      = (const float*)d_in[8];
  const float* wv      = (const float*)d_in[9];
  const float* rq      = (const float*)d_in[10];
  const float* ro      = (const float*)d_in[11];
  const float* ln2     = (const float*)d_in[12];
  const float* Wgf     = (const float*)d_in[13];
  const float* Wuf     = (const float*)d_in[14];
  const float* Wof     = (const float*)d_in[15];
  const float* ln_out  = (const float*)d_in[16];
  float* out = (float*)d_out;

  // workspace carve-up (~211 MB: 129 MB activations + 82 MB bf16 weight pool)
  char* ws = (char*)d_ws;
  size_t off = 0;
  auto carve = [&](size_t bytes) { char* p = ws + off; off = (off + bytes + 255) & ~(size_t)255; return p; };
  float*  x     = (float*) carve((size_t)MROWS * DV * 4);
  __bf16* hbf   = (__bf16*)carve((size_t)MROWS * DV * 2);
  float*  hm    = (float*) carve((size_t)MROWS * DV * 4);
  float*  gv    = (float*) carve((size_t)MROWS * 2 * DV * 4);
  __bf16* cgbf  = (__bf16*)carve((size_t)MROWS * DV * 2);
  float*  gbuf  = (float*) carve((size_t)MROWS * SV * 4);
  float*  vbuf  = (float*) carve((size_t)MROWS * DSV * 4);
  float*  rwbuf = (float*) carve((size_t)MROWS * SV * 4);
  __bf16* rfbf  = (__bf16*)carve((size_t)MROWS * SV * DSV * 2);
  float*  mem0  = (float*) carve((size_t)BV * SV * DSV * 4);
  float*  mem1  = (float*) carve((size_t)BV * SV * DSV * 4);
  float*  gf    = (float*) carve((size_t)MROWS * FFV * 4);
  float*  uf    = (float*) carve((size_t)MROWS * FFV * 4);
  __bf16* upbf  = (__bf16*)carve((size_t)MROWS * FFV * 2);
  // bf16 weight pool
  __bf16* embf  = (__bf16*)carve((size_t)VVOC * DV * 2);
  __bf16* mupbf = (__bf16*)carve((size_t)LV * 2 * DV * DV * 2);
  __bf16* mdbf  = (__bf16*)carve((size_t)LV * DV * DV * 2);
  __bf16* robf  = (__bf16*)carve((size_t)LV * DV * SV * DSV * 2);
  __bf16* Wgfbf = (__bf16*)carve((size_t)LV * FFV * DV * 2);
  __bf16* Wufbf = (__bf16*)carve((size_t)LV * FFV * DV * 2);
  __bf16* Wofbf = (__bf16*)carve((size_t)LV * DV * FFV * 2);
  (void)ws_size; (void)in_sizes; (void)n_in; (void)out_size;

  auto cvt = [&](const float* src, __bf16* dst, size_t n) {
    int n4 = (int)(n / 4);
    hipLaunchKernelGGL(k_cvt_bf16, dim3((n4 + 255) / 256), dim3(256), 0, stream, src, dst, n4);
  };
  cvt(embed, embf, (size_t)VVOC * DV);
  cvt(mup,   mupbf,(size_t)LV * 2 * DV * DV);
  cvt(mdown, mdbf, (size_t)LV * DV * DV);
  cvt(ro,    robf, (size_t)LV * DV * SV * DSV);
  cvt(Wgf,   Wgfbf,(size_t)LV * FFV * DV);
  cvt(Wuf,   Wufbf,(size_t)LV * FFV * DV);
  cvt(Wof,   Wofbf,(size_t)LV * DV * FFV);

  auto gemm = [&](const __bf16* A, const __bf16* W, float* O, int N, int Kd, int mode) {
    dim3 grid(MROWS / 128, (N + 127) / 128);
    if (mode == 1)      hipLaunchKernelGGL(k_gemm_bf16<1>, grid, dim3(256), 0, stream, A, W, O, N, Kd);
    else if (mode == 2) hipLaunchKernelGGL(k_gemm_bf16<2>, grid, dim3(256), 0, stream, A, W, O, N, Kd);
    else                hipLaunchKernelGGL(k_gemm_bf16<0>, grid, dim3(256), 0, stream, A, W, O, N, Kd);
  };

  hipMemsetAsync(mem0, 0, (size_t)BV * SV * DSV * 4, stream);

  // h = rmsnorm(embed[tokens], ln_in) -> residual stream x (f32)
  hipLaunchKernelGGL(k_rmsnorm, dim3(MROWS), dim3(256), 0, stream,
                     embed, tokens, ln_in, x, (__bf16*)nullptr);

  float* memin = mem0;
  float* memout = mem1;
  for (int l = 0; l < LV; ++l) {
    const float*  ln1_l  = ln1   + (size_t)l * DV;
    const __bf16* mup_l  = mupbf + (size_t)l * 2 * DV * DV;
    const float*  cw_l   = convw + (size_t)l * DV * KCONV;
    const __bf16* md_l   = mdbf  + (size_t)l * DV * DV;
    const float*  lnm_l  = lnmem + (size_t)l * DV;
    const float*  wg_l   = wg    + (size_t)l * SV * DV;
    const float*  wv_l   = wv    + (size_t)l * DSV * DV;
    const float*  rq_l   = rq    + (size_t)l * SV * DV;
    const __bf16* ro_l   = robf  + (size_t)l * DV * SV * DSV;
    const float*  ln2_l  = ln2   + (size_t)l * DV;
    const __bf16* Wgf_l  = Wgfbf + (size_t)l * FFV * DV;
    const __bf16* Wuf_l  = Wufbf + (size_t)l * FFV * DV;
    const __bf16* Wof_l  = Wofbf + (size_t)l * DV * FFV;

    // mixer: gv = rmsnorm(x,ln1) @ mixer_up^T ; conv+gate ; x += cg @ md^T
    hipLaunchKernelGGL(k_rmsnorm, dim3(MROWS), dim3(256), 0, stream,
                       x, (const int*)nullptr, ln1_l, (float*)nullptr, hbf);
    gemm(hbf, mup_l, gv, 2 * DV, DV, 0);
    hipLaunchKernelGGL(k_convgate, dim3((MROWS * DV + 255) / 256), dim3(256), 0, stream,
                       gv, cw_l, cgbf);
    gemm(cgbf, md_l, x, DV, DV, 1);

    // story memory: small projections, affine scan, ctx = rf @ ro^T
    hipLaunchKernelGGL(k_rmsnorm, dim3(MROWS), dim3(256), 0, stream,
                       x, (const int*)nullptr, lnm_l, hm, (__bf16*)nullptr);
    hipLaunchKernelGGL(k_smalldense, dim3(MROWS), dim3(256), 0, stream,
                       hm, wg_l, wv_l, rq_l, gbuf, vbuf, rwbuf);
    hipLaunchKernelGGL(k_scan, dim3((BV * SV * DSV) / 8), dim3(256), 0, stream,
                       gbuf, vbuf, rwbuf, memin, memout, rfbf);
    gemm(rfbf, ro_l, x, DV, SV * DSV, 1);

    // FFN: x += (silu(h@Wgf^T) * (h@Wuf^T)) @ Wof^T
    hipLaunchKernelGGL(k_rmsnorm, dim3(MROWS), dim3(256), 0, stream,
                       x, (const int*)nullptr, ln2_l, (float*)nullptr, hbf);
    gemm(hbf, Wgf_l, gf, FFV, DV, 0);
    gemm(hbf, Wuf_l, uf, FFV, DV, 0);
    hipLaunchKernelGGL(k_siluMul, dim3((MROWS * FFV + 255) / 256), dim3(256), 0, stream,
                       gf, uf, upbf, MROWS * FFV);
    gemm(upbf, Wof_l, x, DV, FFV, 1);

    float* tmp = memin; memin = memout; memout = tmp;
  }

  // tied head: logits = rmsnorm(x, ln_out) @ embed^T  (nontemporal stores)
  hipLaunchKernelGGL(k_rmsnorm, dim3(MROWS), dim3(256), 0, stream,
                     x, (const int*)nullptr, ln_out, (float*)nullptr, hbf);
  gemm(hbf, embf, out, VVOC, DV, 2);
}
